// VQVAE_10660108828681
// MI455X (gfx1250) — matrix operand order repaired and checked by reading, not compiled, not run on hardware
//
#include <hip/hip_runtime.h>
#include <hip/hip_bf16.h>

// ---------------------------------------------------------------------------
// VQ-VAE forward for gfx1250 (CDNA5, wave32, WMMA).
// All convs / transposed convs run as implicit-GEMM f16 WMMA with f32 accum.
// Conv geometry is templated; im2col gathers are branchless (clamped+select).
// ---------------------------------------------------------------------------

typedef __attribute__((ext_vector_type(16))) _Float16 v16h;
typedef __attribute__((ext_vector_type(8)))  _Float16 v8h;
typedef __attribute__((ext_vector_type(8)))  float    v8f;

__device__ inline v16h cat16(v8h lo, v8h hi) {
  return __builtin_shufflevector(lo, hi, 0,1,2,3,4,5,6,7,8,9,10,11,12,13,14,15);
}

// --------------------------- weight packing --------------------------------
__global__ void vqvae_pack_w(const float* __restrict__ w, _Float16* __restrict__ wp,
                             int M, int Cin, int KH, int KW, int Mp, int Kp, int flip) {
  int i = blockIdx.x * blockDim.x + threadIdx.x;
  int total = Mp * Kp;
  if (i >= total) return;
  int m = i / Kp, k = i - m * Kp;
  int Kr = Cin * KH * KW;
  _Float16 v = (_Float16)0.0f;
  if (m < M && k < Kr) {
    int ci = k / (KH * KW);
    int r  = k - ci * (KH * KW);
    int ky = r / KW, kx = r - ky * KW;
    if (flip) { ky = KH - 1 - ky; kx = KW - 1 - kx; }
    v = (_Float16)w[((m * Cin + ci) * KH + ky) * KW + kx];
  }
  wp[i] = v;
}

// ----------------------- implicit-GEMM conv (WMMA) -------------------------
// out[M,N] = Wp[Mp,Kp] x Patches[Kp,N] (+bias, optional relu)
// Block: 128 threads = 4 waves; tile = 16 (M) x 256 (N); K-step 32.
template <int KH, int KW, int STRIDE, int TRANS, int LOGW>
__global__ __launch_bounds__(128)
void vqvae_conv_gemm_wmma(const float* __restrict__ in,      // [B][Cin][H][W] f32
                          const _Float16* __restrict__ wp,    // [Mp][Kp] f16
                          const float* __restrict__ bias,     // [Mreal]
                          float* __restrict__ out,            // [B][Mreal][Ho][Wo] f32
                          int Bn, int Cin, int H, int W, int pad,
                          int Mreal, int Mp, int Kp, int relu) {
  __shared__ __align__(16) _Float16 Bs[256][40];   // [n][k] tile, 80B row stride

  constexpr int Wo  = 1 << LOGW;
  constexpr int HWo = 1 << (2 * LOGW);
  constexpr int KHW = KH * KW;

  const int tid  = threadIdx.x;
  const int lane = tid & 31;
  const int wave = tid >> 5;
  const int n0   = blockIdx.x * 256;
  const int m0   = blockIdx.y * 16;
  const int Kreal = Cin * KHW;
  const int HWin  = H * W;

  // Per-thread output-pixel decode for the two LDS rows this thread fills.
  int rows[2] = { tid, tid + 128 };
  int poy[2], pox[2];
  const float* inb[2];
#pragma unroll
  for (int t = 0; t < 2; ++t) {
    int n   = n0 + rows[t];
    int b   = n >> (2 * LOGW);
    int rem = n & (HWo - 1);
    poy[t]  = rem >> LOGW;
    pox[t]  = rem & (Wo - 1);
    inb[t]  = in + (size_t)b * Cin * HWin;
  }

  v8f acc[4] = {};

  const int mA  = m0 + (lane & 15);
  const int kbA = (lane >> 4) * 8;   // f16 A layout: K 0-7/16-23 vs 8-15/24-31
  const int kbB = (lane >> 4) * 8;   // f16 B layout: K 0-7/16-23 vs 8-15/24-31

  for (int k0 = 0; k0 < Kp; k0 += 32) {
    // ---- cooperative implicit-im2col fill of B tile (32 K x 256 N) ----
    // Branchless: clamp offset with a select, load unconditionally, select 0.
#pragma unroll
    for (int t = 0; t < 2; ++t) {
      const int row = rows[t];
      const float* __restrict__ base = inb[t];
#pragma unroll
      for (int c8 = 0; c8 < 4; ++c8) {
        v8h packv;
#pragma unroll
        for (int e = 0; e < 8; ++e) {
          int k  = k0 + c8 * 8 + e;
          int ci = k / KHW;            // constexpr divisor -> shifts/mulhi
          int r  = k - ci * KHW;
          int ky = r / KW, kx = r - ky * KW;
          int iy, ix; bool ok;
          if (!TRANS) {
            iy = poy[t] * STRIDE - pad + ky;
            ix = pox[t] * STRIDE - pad + kx;
            ok = ((unsigned)iy < (unsigned)H) & ((unsigned)ix < (unsigned)W);
          } else {                     // lhs-dilated conv, kernel pre-flipped
            int iyd = poy[t] - (KH - 1 - pad) + ky;
            int ixd = pox[t] - (KW - 1 - pad) + kx;
            if (STRIDE == 1) {
              iy = iyd; ix = ixd;
              ok = ((unsigned)iy < (unsigned)H) & ((unsigned)ix < (unsigned)W);
            } else {
              ok = (iyd >= 0) & (ixd >= 0) &
                   ((iyd & (STRIDE - 1)) == 0) & ((ixd & (STRIDE - 1)) == 0);
              iy = iyd >> 1; ix = ixd >> 1;   // STRIDE == 2
              ok = ok & (iy < H) & (ix < W);
            }
          }
          ok = ok & (k < Kreal);
          int off = ci * HWin + iy * W + ix;
          off = ok ? off : 0;               // v_cndmask, keeps load in-bounds
          float raw = base[off];            // unconditional global_load_b32
          packv[e] = ok ? (_Float16)raw : (_Float16)0.0f;
        }
        *(v8h*)&Bs[row][c8 * 8] = packv;    // one ds_store_b128 per 8 halves
      }
    }
    __syncthreads();

    // ---- A fragment: two b128 loads from packed weights ----
    const _Float16* ap = wp + (size_t)mA * Kp + k0 + kbA;
    __builtin_prefetch(ap + 32, 0, 1);      // global_prefetch_b8 next K-chunk
    v8h alo = *(const v8h*)ap;
    v8h ahi = *(const v8h*)(ap + 16);
    v16h afrag = cat16(alo, ahi);

    // ---- 4 x 16x16x32 WMMA per wave ----
#pragma unroll
    for (int j = 0; j < 4; ++j) {
      int nl = wave * 64 + j * 16 + (lane & 15);
      const _Float16* bp = &Bs[nl][kbB];
      v8h blo = *(const v8h*)bp;
      v8h bhi = *(const v8h*)(bp + 16);
      v16h bfrag = cat16(blo, bhi);
      acc[j] = __builtin_amdgcn_wmma_f32_16x16x32_f16(
          false, afrag, false, bfrag, (short)0, acc[j], false, false);
    }
    __syncthreads();
  }

  // ---- epilogue: bias (+relu), scatter to NCHW ----
#pragma unroll
  for (int j = 0; j < 4; ++j) {
    int n   = n0 + wave * 64 + j * 16 + (lane & 15);
    int b   = n >> (2 * LOGW);
    int rem = n & (HWo - 1);
    int oy  = rem >> LOGW;
    int ox  = rem & (Wo - 1);
#pragma unroll
    for (int r = 0; r < 8; ++r) {
      int m = m0 + r + ((lane >> 4) << 3);
      if (m < Mreal) {
        float v = acc[j][r] + bias[m];
        if (relu) v = fmaxf(v, 0.0f);
        out[(((size_t)b * Mreal + m) << (2 * LOGW)) + (oy << LOGW) + ox] = v;
      }
    }
  }
}

// ----------------------------- batch norm ----------------------------------
__global__ __launch_bounds__(256)
void vqvae_bn_stats(const float* __restrict__ x, float* __restrict__ mean,
                    float* __restrict__ var, int C, int Bn, int HW) {
  int c = blockIdx.x;
  float s = 0.f, s2 = 0.f;
  for (int b = 0; b < Bn; ++b) {
    const float* xp = x + ((size_t)b * C + c) * HW;
    for (int j = threadIdx.x; j < HW; j += 256) {
      float v = xp[j];
      s += v; s2 += v * v;
    }
  }
  __shared__ float ss[256], sq[256];
  ss[threadIdx.x] = s; sq[threadIdx.x] = s2;
  __syncthreads();
  for (int st = 128; st > 0; st >>= 1) {
    if ((int)threadIdx.x < st) {
      ss[threadIdx.x] += ss[threadIdx.x + st];
      sq[threadIdx.x] += sq[threadIdx.x + st];
    }
    __syncthreads();
  }
  if (threadIdx.x == 0) {
    float inv = 1.0f / (float)(Bn * HW);
    float m = ss[0] * inv;
    mean[c] = m;
    var[c]  = sq[0] * inv - m * m;
  }
}

// c = (i >> logHW) & Cmask   (C and HW are powers of two everywhere here)
__global__ void vqvae_bn_apply(const float* __restrict__ x, const float* __restrict__ mean,
                               const float* __restrict__ var, const float* __restrict__ g,
                               const float* __restrict__ b, float* __restrict__ out,
                               int logHW, int Cmask, int n, int relu) {
  int i = blockIdx.x * blockDim.x + threadIdx.x;
  if (i >= n) return;
  int c = (i >> logHW) & Cmask;
  float v = (x[i] - mean[c]) * rsqrtf(var[c] + 1e-5f) * g[c] + b[c];
  if (relu) v = fmaxf(v, 0.0f);
  out[i] = v;
}

// ----------------------------- elementwise ---------------------------------
__global__ void vqvae_relu(const float* __restrict__ x, float* __restrict__ o, int n) {
  int i = blockIdx.x * blockDim.x + threadIdx.x;
  if (i < n) o[i] = fmaxf(x[i], 0.0f);
}
__global__ void vqvae_add(const float* __restrict__ x, const float* __restrict__ y,
                          float* __restrict__ o, int n) {
  int i = blockIdx.x * blockDim.x + threadIdx.x;
  if (i < n) o[i] = x[i] + y[i];
}
__global__ void vqvae_tanh(const float* __restrict__ x, float* __restrict__ o, int n) {
  int i = blockIdx.x * blockDim.x + threadIdx.x;
  if (i < n) o[i] = tanhf(x[i]);
}
__global__ void vqvae_zero(float* __restrict__ p, int n) {
  int i = blockIdx.x * blockDim.x + threadIdx.x;
  if (i < n) p[i] = 0.0f;
}

// ------------------------------- VQ ----------------------------------------
// z: (B,64,1024); E: (512,64). One thread per pixel, 64-dim vector in regs.
__global__ __launch_bounds__(128)
void vqvae_vq_argmin(const float* __restrict__ z, const float* __restrict__ E,
                     int* __restrict__ idx, int Bn) {
  int n = blockIdx.x * blockDim.x + threadIdx.x;
  if (n >= (Bn << 10)) return;
  int b = n >> 10, j = n & 1023;
  float zv[64];
#pragma unroll
  for (int c = 0; c < 64; ++c) zv[c] = z[(((size_t)b << 6) + c) * 1024 + j];
  float best = 3.4e38f;
  int bi = 0;
  for (int e = 0; e < 512; ++e) {
    const float* ep = E + e * 64;
    float d = 0.f;
#pragma unroll
    for (int c = 0; c < 64; ++c) { float t = zv[c] - ep[c]; d = fmaf(t, t, d); }
    if (d < best) { best = d; bi = e; }
  }
  idx[n] = bi;
}

// latents (forward of straight-through) = zq; qloss = (1+beta)*mean((zq-z)^2)
__global__ void vqvae_vq_out(const float* __restrict__ z, const float* __restrict__ E,
                             const int* __restrict__ idx, float* __restrict__ latents,
                             float* __restrict__ loss, int Bn) {
  int i = blockIdx.x * blockDim.x + threadIdx.x;
  int total = Bn << 16;                // B * 64 * 1024
  if (i >= total) return;
  int b   = i >> 16;
  int rem = i & 65535;
  int c   = rem >> 10;
  int j   = rem & 1023;
  int n   = (b << 10) | j;
  float zq = E[idx[n] * 64 + c];
  latents[i] = zq;
  float d = zq - z[i];
  atomicAdd(loss, (1.0f + 0.25f) * d * d / (float)total);
}

// ---------------------------------------------------------------------------
extern "C" void kernel_launch(void* const* d_in, const int* in_sizes, int n_in,
                              void* d_out, int out_size, void* d_ws, size_t ws_size,
                              hipStream_t stream) {
  (void)in_sizes; (void)n_in; (void)out_size; (void)ws_size;
  // jax pytree flatten order (dict keys sorted, depth-first):
  const float* codebook = (const float*)d_in[0];   // (512,64)
  const float* db1   = (const float*)d_in[1];
  const float* db2   = (const float*)d_in[2];
  const float* db3   = (const float*)d_in[3];
  const float* dbn2b = (const float*)d_in[4];
  const float* dbn2g = (const float*)d_in[5];
  const float* dr_b1  = (const float*)d_in[6];
  const float* dr_b2  = (const float*)d_in[7];
  const float* dr_bb1 = (const float*)d_in[8];
  const float* dr_bb2 = (const float*)d_in[9];
  const float* dr_g1  = (const float*)d_in[10];
  const float* dr_g2  = (const float*)d_in[11];
  const float* dr_w1  = (const float*)d_in[12];    // (128,128,3,3)
  const float* dr_w2  = (const float*)d_in[13];    // (128,128,1,1)
  const float* dw1   = (const float*)d_in[14];     // (128,128,3,3)
  const float* dw2   = (const float*)d_in[15];     // (64,128,4,4)
  const float* dw3   = (const float*)d_in[16];     // (3,64,4,4)
  const float* eb1   = (const float*)d_in[17];
  const float* eb2   = (const float*)d_in[18];
  const float* eb3   = (const float*)d_in[19];
  const float* ebn1b = (const float*)d_in[20];
  const float* ebn1g = (const float*)d_in[21];
  const float* er_b1  = (const float*)d_in[22];
  const float* er_b2  = (const float*)d_in[23];
  const float* er_bb1 = (const float*)d_in[24];
  const float* er_bb2 = (const float*)d_in[25];
  const float* er_g1  = (const float*)d_in[26];
  const float* er_g2  = (const float*)d_in[27];
  const float* er_w1  = (const float*)d_in[28];    // (128,128,3,3)
  const float* er_w2  = (const float*)d_in[29];    // (128,128,1,1)
  const float* ew1   = (const float*)d_in[30];     // (64,3,4,4)
  const float* ew2   = (const float*)d_in[31];     // (128,64,4,4)
  const float* ew3   = (const float*)d_in[32];     // (128,128,3,3)
  const float* pb    = (const float*)d_in[33];
  const float* pw    = (const float*)d_in[34];     // (64,128,1,1)
  const float* qb    = (const float*)d_in[35];
  const float* qw    = (const float*)d_in[36];     // (128,64,1,1)
  const float* x     = (const float*)d_in[37];     // (32,3,128,128)

  const int Bn = 32;
  float* outp = (float*)d_out;
  const int ENC_N  = 32 * 128 * 32 * 32;   // 4194304
  const int DEC_N  = 32 * 3 * 128 * 128;   // 1572864
  const int LAT_N  = 32 * 64 * 32 * 32;    // 2097152
  float* enc_out  = outp;
  float* dec_out  = outp + ENC_N;
  float* lat_out  = outp + ENC_N + DEC_N;
  float* loss_out = outp + ENC_N + DEC_N + LAT_N;

  // ---- workspace bump allocator ----
  char* ws = (char*)d_ws;
  size_t off = 0;
  auto alloc = [&](size_t bytes) -> void* {
    off = (off + 255) & ~(size_t)255;
    void* p = ws + off;
    off += bytes;
    return p;
  };

  // packed f16 weights (Mp x Kp)
  _Float16* wp_e1  = (_Float16*)alloc((size_t)64  * 64   * 2);
  _Float16* wp_e2  = (_Float16*)alloc((size_t)128 * 1024 * 2);
  _Float16* wp_e3  = (_Float16*)alloc((size_t)128 * 1152 * 2);
  _Float16* wp_er1 = (_Float16*)alloc((size_t)128 * 1152 * 2);
  _Float16* wp_er2 = (_Float16*)alloc((size_t)128 * 128  * 2);
  _Float16* wp_pw  = (_Float16*)alloc((size_t)64  * 128  * 2);
  _Float16* wp_qw  = (_Float16*)alloc((size_t)128 * 64   * 2);
  _Float16* wp_d1  = (_Float16*)alloc((size_t)128 * 1152 * 2);
  _Float16* wp_dr1 = (_Float16*)alloc((size_t)128 * 1152 * 2);
  _Float16* wp_dr2 = (_Float16*)alloc((size_t)128 * 128  * 2);
  _Float16* wp_d2  = (_Float16*)alloc((size_t)64  * 2048 * 2);
  _Float16* wp_d3  = (_Float16*)alloc((size_t)16  * 1024 * 2);

  float* bufA = (float*)alloc((size_t)8388608 * 4);   // 32x64x64x64
  float* bufC = (float*)alloc((size_t)4194304 * 4);   // 32x128x32x32
  float* bufD = (float*)alloc((size_t)4194304 * 4);
  float* bufE = (float*)alloc((size_t)4194304 * 4);
  float* bufF = (float*)alloc((size_t)4194304 * 4);
  float* bufG = (float*)alloc((size_t)4194304 * 4);
  float* bufZ = (float*)alloc((size_t)2097152 * 4);   // 32x64x32x32
  float* bufDec = (float*)alloc((size_t)1572864 * 4); // 32x3x128x128
  float* meanb = (float*)alloc(128 * 4);
  float* varb  = (float*)alloc(128 * 4);
  int*   idxb  = (int*)alloc(32768 * 4);

  auto pack = [&](const float* w, _Float16* wp, int M, int Cin, int KH, int KW,
                  int Mp, int Kp, int flip) {
    int n = Mp * Kp;
    vqvae_pack_w<<<dim3((n + 255) / 256), dim3(256), 0, stream>>>(
        w, wp, M, Cin, KH, KW, Mp, Kp, flip);
  };
  auto bn = [&](float* xio, const float* g, const float* b, int C, int HW, int logHW,
                int relu) {
    int n = Bn * C * HW;
    vqvae_bn_stats<<<dim3(C), dim3(256), 0, stream>>>(xio, meanb, varb, C, Bn, HW);
    vqvae_bn_apply<<<dim3((n + 255) / 256), dim3(256), 0, stream>>>(
        xio, meanb, varb, g, b, xio, logHW, C - 1, n, relu);
  };

// CONV(KH,KW,STRIDE,TRANS,LOGWo, in, w, bias, out, Cin,H,W,pad, M,Mp,Kp, relu)
#define CONV(KHv, KWv, Sv, Tv, LGv, inp, wgt, bia, op, Cin, H, W, pad, M, Mp, Kp, relu)   \
  vqvae_conv_gemm_wmma<KHv, KWv, Sv, Tv, LGv>                                             \
      <<<dim3((Bn << (2 * LGv)) / 256, (Mp) / 16), dim3(128), 0, stream>>>(               \
          inp, wgt, bia, op, Bn, Cin, H, W, pad, M, Mp, Kp, relu)

  // ---- pack weights (flip=1 for transposed convs) ----
  pack(ew1,   wp_e1,  64, 3, 4, 4,   64, 64,   0);
  pack(ew2,   wp_e2,  128, 64, 4, 4, 128, 1024, 0);
  pack(ew3,   wp_e3,  128, 128, 3, 3, 128, 1152, 0);
  pack(er_w1, wp_er1, 128, 128, 3, 3, 128, 1152, 0);
  pack(er_w2, wp_er2, 128, 128, 1, 1, 128, 128,  0);
  pack(pw,    wp_pw,  64, 128, 1, 1,  64, 128,  0);
  pack(qw,    wp_qw,  128, 64, 1, 1, 128, 64,   0);
  pack(dw1,   wp_d1,  128, 128, 3, 3, 128, 1152, 1);
  pack(dr_w1, wp_dr1, 128, 128, 3, 3, 128, 1152, 0);
  pack(dr_w2, wp_dr2, 128, 128, 1, 1, 128, 128,  0);
  pack(dw2,   wp_d2,  64, 128, 4, 4,  64, 2048, 1);
  pack(dw3,   wp_d3,  3, 64, 4, 4,    16, 1024, 1);

  // ---- encoder ----
  CONV(4,4,2,0,6, x,    wp_e1, eb1, bufA, 3,128,128,1,  64, 64, 64,   0);
  bn(bufA, ebn1g, ebn1b, 64, 4096, 12, 1);                                // h1
  CONV(4,4,2,0,5, bufA, wp_e2, eb2, bufC, 64,64,64,1,  128,128,1024, 1);  // h2 (relu fused)
  CONV(3,3,1,0,5, bufC, wp_e3, eb3, bufD, 128,32,32,1, 128,128,1152, 0);  // h

  for (int it = 0; it < 2; ++it) {   // weight-shared encoder resblocks
    vqvae_relu<<<dim3(ENC_N / 256), dim3(256), 0, stream>>>(bufD, bufE, ENC_N);
    CONV(3,3,1,0,5, bufE, wp_er1, er_b1, bufF, 128,32,32,1, 128,128,1152, 0);
    bn(bufF, er_g1, er_bb1, 128, 1024, 10, 1);
    CONV(1,1,1,0,5, bufF, wp_er2, er_b2, bufG, 128,32,32,0, 128,128,128,  0);
    bn(bufG, er_g2, er_bb2, 128, 1024, 10, 0);
    vqvae_add<<<dim3(ENC_N / 256), dim3(256), 0, stream>>>(bufD, bufG, bufD, ENC_N);
  }
  vqvae_relu<<<dim3(ENC_N / 256), dim3(256), 0, stream>>>(bufD, enc_out, ENC_N);  // encoded

  // ---- VQ bottleneck ----
  CONV(1,1,1,0,5, enc_out, wp_pw, pb, bufZ, 128,32,32,0, 64,64,128, 0);   // z
  vqvae_vq_argmin<<<dim3(32768 / 128), dim3(128), 0, stream>>>(bufZ, codebook, idxb, Bn);
  vqvae_zero<<<dim3(1), dim3(1), 0, stream>>>(loss_out, 1);
  vqvae_vq_out<<<dim3(LAT_N / 256), dim3(256), 0, stream>>>(
      bufZ, codebook, idxb, lat_out, loss_out, Bn);

  // ---- decoder ----
  CONV(1,1,1,0,5, lat_out, wp_qw, qb, bufF, 64,32,32,0, 128,128,64, 0);   // d0
  CONV(3,3,1,1,5, bufF, wp_d1, db1, bufD, 128,32,32,1, 128,128,1152, 0);  // d1 (convT s=1)

  for (int it = 0; it < 2; ++it) {   // weight-shared decoder resblocks
    vqvae_relu<<<dim3(ENC_N / 256), dim3(256), 0, stream>>>(bufD, bufE, ENC_N);
    CONV(3,3,1,0,5, bufE, wp_dr1, dr_b1, bufF, 128,32,32,1, 128,128,1152, 0);
    bn(bufF, dr_g1, dr_bb1, 128, 1024, 10, 1);
    CONV(1,1,1,0,5, bufF, wp_dr2, dr_b2, bufG, 128,32,32,0, 128,128,128,  0);
    bn(bufG, dr_g2, dr_bb2, 128, 1024, 10, 0);
    vqvae_add<<<dim3(ENC_N / 256), dim3(256), 0, stream>>>(bufD, bufG, bufD, ENC_N);
  }
  vqvae_relu<<<dim3(ENC_N / 256), dim3(256), 0, stream>>>(bufD, bufE, ENC_N);
  CONV(4,4,2,1,6, bufE, wp_d2, db2, bufA, 128,32,32,1,  64,64,2048, 0);   // convT s=2
  bn(bufA, dbn2g, dbn2b, 64, 4096, 12, 1);
  CONV(4,4,2,1,7, bufA, wp_d3, db3, bufDec, 64,64,64,1, 3,16,1024, 0);    // convT s=2
  vqvae_tanh<<<dim3(DEC_N / 256), dim3(256), 0, stream>>>(bufDec, dec_out, DEC_N);

#undef CONV
}
